// IS_FTT_multi_layer_relu_l_61512521613619
// MI455X (gfx1250) — compile-verified, matrix operand order checked by reading, not run on hardware
//
#include <hip/hip_runtime.h>

typedef __attribute__((ext_vector_type(16))) __bf16 v16bf;
typedef __attribute__((ext_vector_type(8)))  __bf16 v8bf;
typedef __attribute__((ext_vector_type(8)))  float  v8f;

union Frag { v16bf v; v8bf h[2]; };

struct TTShape { int nd[5]; int md[5]; };

// ---------------------------------------------------------------------------
// Kernel 1: expand TT cores -> dense bf16 weight matrix Wt[Npad][Kpad],
// K contiguous (i.e. transposed weight), zero padded.
// ---------------------------------------------------------------------------
__global__ void tt_densify(const float* __restrict__ g0, const float* __restrict__ g1,
                           const float* __restrict__ g2, const float* __restrict__ g3,
                           const float* __restrict__ g4,
                           TTShape s, int Kpad, int Npad, int Kreal, int Nreal,
                           __bf16* __restrict__ outw) {
    long t = (long)blockIdx.x * 256 + threadIdx.x;
    long total = (long)Kpad * Npad;
    if (t >= total) return;
    int n = (int)(t / Kpad);
    int k = (int)(t % Kpad);
    float val = 0.f;
    if (n < Nreal && k < Kreal) {
        int ni[5], mi[5];
        int kk = k;
        #pragma unroll
        for (int d = 4; d >= 0; --d) { ni[d] = kk % s.nd[d]; kk /= s.nd[d]; }
        int nn = n;
        #pragma unroll
        for (int d = 4; d >= 0; --d) { mi[d] = nn % s.md[d]; nn /= s.md[d]; }
        const int rk[6] = {1, 3, 3, 3, 3, 1};
        const float* G[5] = {g0, g1, g2, g3, g4};
        float v[3], w[3];
        for (int r = 0; r < rk[1]; ++r)
            v[r] = g0[(ni[0] * s.md[0] + mi[0]) * rk[1] + r];
        #pragma unroll
        for (int st = 1; st < 5; ++st) {
            const float* g = G[st];
            for (int rn = 0; rn < rk[st + 1]; ++rn) {
                float acc = 0.f;
                for (int rp = 0; rp < rk[st]; ++rp)
                    acc += v[rp] * g[((rp * s.nd[st] + ni[st]) * s.md[st] + mi[st]) * rk[st + 1] + rn];
                w[rn] = acc;
            }
            for (int r = 0; r < rk[st + 1]; ++r) v[r] = w[r];
        }
        val = v[0];
    }
    outw[t] = (__bf16)val;
}

// ---------------------------------------------------------------------------
// Fused MLP kernel
// ---------------------------------------------------------------------------
#define RB_A 6144   // row bytes for 3072-wide bf16 buffer
#define RB_H 4096   // row bytes for 2048-wide bf16 buffer
#define LDS_BYTES (32 * RB_A + 32 * RB_H)   // 196608 + 131072 = 327680 = 320 KB

// XOR-swizzled byte offset: 16B chunks within a row are permuted by row%16 so
// A-fragment ds_load_b128 across 16 rows hits distinct bank groups.
__device__ __forceinline__ int swz(int row, int col, int rowBytes) {
    int c  = col >> 3;                 // 16B chunk index (8 bf16 per chunk)
    int ph = c ^ (row & 15);
    return row * rowBytes + (ph << 4) + ((col & 7) << 1);
}

// A fragment (16x32 bf16): lane<16 -> M=lane, K chunks [kb..kb+8) and [kb+16..kb+24)
//                          lane>=16 -> M=lane-16, K chunks +8 / +24
__device__ __forceinline__ v16bf load_a(const char* buf, int rowBytes, int mbase,
                                        int kb, int lane) {
    int  lq = lane & 15;
    int  k0 = kb + ((lane < 16) ? 0 : 8);
    int  m  = mbase + lq;
    Frag f;
    f.h[0] = *(const v8bf*)(buf + swz(m, k0,      rowBytes));
    f.h[1] = *(const v8bf*)(buf + swz(m, k0 + 16, rowBytes));
    return f.v;
}

// B fragment (32x16 bf16), weights stored [N][K] so K is contiguous per lane.
__device__ __forceinline__ v16bf load_b(const __bf16* __restrict__ Wt, int Kpad,
                                        int n, int kb, int lane) {
    const __bf16* p = Wt + (size_t)n * Kpad + kb + ((lane < 16) ? 0 : 8);
    Frag f;
    f.h[0] = *(const v8bf*)(p);
    f.h[1] = *(const v8bf*)(p + 16);
    return f.v;
}

__device__ __forceinline__ v8f wmma_bf16(v16bf a, v16bf b, v8f c) {
    return __builtin_amdgcn_wmma_f32_16x16x32_bf16(false, a, false, b,
                                                   (short)0, c, false, false);
}

// One dense layer: [32 x Kpad] (LDS, bf16, swizzled) x Wt[2048 x Kpad] -> [32 x 2048]
// 8 waves, each owns 16 N-tiles, processed in 4 chunks of (2 M-tiles x 4 N-tiles).
__device__ __forceinline__ void gemm_layer(const __bf16* __restrict__ Wt, int Kpad,
                                           const char* inb, int inRB,
                                           char* outb, int outRB,
                                           int tid, bool relu) {
    int  wave = tid >> 5;
    int  lane = tid & 31;
    int  lq   = lane & 15;
    bool lo   = lane < 16;
    for (int g = 0; g < 4; ++g) {
        int ntb = (wave * 16 + g * 4) * 16;   // base output column
        v8f acc[2][4];
        v8f zero = {0.f, 0.f, 0.f, 0.f, 0.f, 0.f, 0.f, 0.f};
        #pragma unroll
        for (int mt = 0; mt < 2; ++mt)
            #pragma unroll
            for (int j = 0; j < 4; ++j) acc[mt][j] = zero;

        for (int kb = 0; kb < Kpad; kb += 32) {
            v16bf a0 = load_a(inb, inRB, 0,  kb, lane);
            v16bf a1 = load_a(inb, inRB, 16, kb, lane);
            #pragma unroll
            for (int j = 0; j < 4; ++j) {
                v16bf b = load_b(Wt, Kpad, ntb + j * 16 + lq, kb, lane);
                acc[0][j] = wmma_bf16(a0, b, acc[0][j]);
                acc[1][j] = wmma_bf16(a1, b, acc[1][j]);
            }
        }
        // Epilogue: C layout -> lanes 0-15: N=lane, M=r ; lanes 16-31: N=lane-16, M=r+8
        #pragma unroll
        for (int mt = 0; mt < 2; ++mt) {
            #pragma unroll
            for (int j = 0; j < 4; ++j) {
                int n = ntb + j * 16 + lq;
                #pragma unroll
                for (int r = 0; r < 8; ++r) {
                    int   m = mt * 16 + (lo ? r : r + 8);
                    float v = acc[mt][j][r];
                    if (relu) v = fmaxf(v, 0.f);
                    *(__bf16*)(outb + swz(m, n, outRB)) = (__bf16)v;
                }
            }
        }
    }
}

__global__ __launch_bounds__(256, 1)
void tt_mlp_fused(const float* __restrict__ x,
                  const __bf16* __restrict__ W1, const __bf16* __restrict__ W2,
                  const __bf16* __restrict__ W3, const __bf16* __restrict__ W4,
                  const __bf16* __restrict__ W5, const __bf16* __restrict__ W6,
                  float* __restrict__ out) {
    extern __shared__ char smem[];
    char* bufA = smem;                 // 192 KB : h0 (32x3072) / later 2048-wide ping
    char* bufB = smem + 32 * RB_A;     // 128 KB : 2048-wide pong (also x staging)
    const int tid     = threadIdx.x;
    const int rowbase = blockIdx.x * 32;

    // -------- phase 0: stage x tile into LDS (bufB is free right now) -------
    float* xl = (float*)bufB;          // [32][180] f32
    for (int i = tid; i < 32 * 177; i += 256) {
        int r = i / 177, j = i % 177;
        xl[r * 180 + j] = x[(size_t)(rowbase + r) * 177 + j];
    }
    __syncthreads();

    // -------- phase 1: TT input densification -> bufA bf16 [32][3072] -------
    for (int e = tid; e < 32 * 3072; e += 256) {
        int row = e / 3072, idx = e - row * 3072;
        const float* xr = xl + row * 180;
        int n1  = idx >> 10;
        int rem = idx & 1023;
        int n2  = rem >> 8;
        int n3  = (rem >> 5) & 7;
        int n4  = (rem >> 3) & 3;
        int n5  = rem & 7;
        float a0 = xr[n1 * 3 + 0], a1 = xr[n1 * 3 + 1], a2 = xr[n1 * 3 + 2];
        const float* c2 = xr + 9 + n2 * 3;      // [r1*12 + r2]
        float b0 = a0 * c2[0] + a1 * c2[12] + a2 * c2[24];
        float b1 = a0 * c2[1] + a1 * c2[13] + a2 * c2[25];
        float b2 = a0 * c2[2] + a1 * c2[14] + a2 * c2[26];
        const float* c3 = xr + 45 + n3 * 3;     // [r2*24 + r3]
        float d0 = b0 * c3[0] + b1 * c3[24] + b2 * c3[48];
        float d1 = b0 * c3[1] + b1 * c3[25] + b2 * c3[49];
        float d2 = b0 * c3[2] + b1 * c3[26] + b2 * c3[50];
        const float* c4 = xr + 117 + n4 * 3;    // [r3*12 + r4]
        float e0 = d0 * c4[0] + d1 * c4[12] + d2 * c4[24];
        float e1 = d0 * c4[1] + d1 * c4[13] + d2 * c4[25];
        float e2 = d0 * c4[2] + d1 * c4[14] + d2 * c4[26];
        const float* c5 = xr + 153 + n5;        // [r4*8]
        float h = e0 * c5[0] + e1 * c5[8] + e2 * c5[16];
        *(__bf16*)(bufA + swz(row, idx, RB_A)) = (__bf16)h;
    }
    __syncthreads();

    // -------- 5 hidden layers, ping-pong A <-> B ---------------------------
    gemm_layer(W1, 3072, bufA, RB_A, bufB, RB_H, tid, true);
    __syncthreads();
    gemm_layer(W2, 2048, bufB, RB_H, bufA, RB_H, tid, true);
    __syncthreads();
    gemm_layer(W3, 2048, bufA, RB_H, bufB, RB_H, tid, true);
    __syncthreads();
    gemm_layer(W4, 2048, bufB, RB_H, bufA, RB_H, tid, true);
    __syncthreads();
    gemm_layer(W5, 2048, bufA, RB_H, bufB, RB_H, tid, true);
    __syncthreads();

    // -------- layer 6 (2048 -> 16) on wave 0, logits f32 into bufA ---------
    if (tid < 32) {
        int  lane = tid;
        int  lq   = lane & 15;
        bool lo   = lane < 16;
        v8f acc0 = {0.f,0.f,0.f,0.f,0.f,0.f,0.f,0.f};
        v8f acc1 = {0.f,0.f,0.f,0.f,0.f,0.f,0.f,0.f};
        for (int kb = 0; kb < 2048; kb += 32) {
            v16bf a0 = load_a(bufB, RB_H, 0,  kb, lane);
            v16bf a1 = load_a(bufB, RB_H, 16, kb, lane);
            v16bf b  = load_b(W6, 2048, lq, kb, lane);
            acc0 = wmma_bf16(a0, b, acc0);
            acc1 = wmma_bf16(a1, b, acc1);
        }
        float* lg = (float*)bufA;              // [32][16] logits
        #pragma unroll
        for (int r = 0; r < 8; ++r) {
            int m = lo ? r : r + 8;
            lg[m * 16 + lq]        = acc0[r];
            lg[(16 + m) * 16 + lq] = acc1[r];
        }
    }
    __syncthreads();

    // -------- log_softmax + writeout (one thread per row) ------------------
    if (tid < 32) {
        const float* lr = (float*)bufA + tid * 16;
        float mx = lr[0];
        for (int j = 1; j < 10; ++j) mx = fmaxf(mx, lr[j]);
        float s = 0.f;
        for (int j = 0; j < 10; ++j) s += expf(lr[j] - mx);
        float lse = mx + logf(s);
        float* o = out + (size_t)(rowbase + tid) * 10;
        for (int j = 0; j < 10; ++j) o[j] = lr[j] - lse;
    }
}

// ---------------------------------------------------------------------------
// Launch
// ---------------------------------------------------------------------------
extern "C" void kernel_launch(void* const* d_in, const int* in_sizes, int n_in,
                              void* d_out, int out_size, void* d_ws, size_t ws_size,
                              hipStream_t stream) {
    const float* x = (const float*)d_in[0];
    const float* cores[6][5];
    for (int l = 0; l < 6; ++l)
        for (int k = 0; k < 5; ++k)
            cores[l][k] = (const float*)d_in[1 + l * 5 + k];

    // ws layout (bf16 dense transposed weights, zero padded)
    char* ws = (char*)d_ws;
    __bf16* W1 = (__bf16*)(ws + 0);                       // 2048 x 3072 = 12,582,912 B
    __bf16* W2 = (__bf16*)(ws + 12582912);                // 2048 x 2048 =  8,388,608 B
    __bf16* W3 = (__bf16*)(ws + 20971520);
    __bf16* W4 = (__bf16*)(ws + 29360128);
    __bf16* W5 = (__bf16*)(ws + 37748736);
    __bf16* W6 = (__bf16*)(ws + 46137344);                // 16 x 2048 = 65,536 B

    TTShape s1 = {{3, 4, 8, 4, 8},   {1, 2, 50, 20, 1}};
    TTShape sm = {{1, 2, 50, 20, 1}, {1, 2, 50, 20, 1}};
    TTShape s6 = {{1, 2, 50, 20, 1}, {1, 2, 5, 1, 1}};

    dim3 blk(256);
    tt_densify<<<(2048 * 3072 + 255) / 256, blk, 0, stream>>>(
        cores[0][0], cores[0][1], cores[0][2], cores[0][3], cores[0][4],
        s1, 3072, 2048, 3072, 2000, W1);
    __bf16* Wmid[4] = {W2, W3, W4, W5};
    for (int l = 0; l < 4; ++l) {
        tt_densify<<<(2048 * 2048 + 255) / 256, blk, 0, stream>>>(
            cores[1 + l][0], cores[1 + l][1], cores[1 + l][2], cores[1 + l][3],
            cores[1 + l][4], sm, 2048, 2048, 2000, 2000, Wmid[l]);
    }
    tt_densify<<<(16 * 2048 + 255) / 256, blk, 0, stream>>>(
        cores[5][0], cores[5][1], cores[5][2], cores[5][3], cores[5][4],
        s6, 2048, 16, 2000, 10, W6);

    int rows = in_sizes[0] / 177;          // 16384
    int grid = rows / 32;                  // 512 workgroups
    tt_mlp_fused<<<grid, blk, LDS_BYTES, stream>>>(
        x, W1, W2, W3, W4, W5, W6, (float*)d_out);
}
